// ErnieImageAttention_52604759441925
// MI455X (gfx1250) — compile-verified, matrix operand order checked by reading, not run on hardware
//
#include <hip/hip_runtime.h>

typedef __attribute__((ext_vector_type(16))) __bf16 v16bf;
typedef __attribute__((ext_vector_type(8)))  float  v8f;

#define S_LEN  4096
#define DMODEL 2048
#define NHEADS 16
#define HDIM   128
#define KB     64      // K-depth staged per GEMM iteration
#define KSTR   72      // staging LDS stride (64 + 8 pad)
#define TSTR   136     // tile   LDS stride (128 + 8 pad)

__device__ inline v8f vzero8() {
  v8f v;
#pragma unroll
  for (int i = 0; i < 8; ++i) v[i] = 0.f;
  return v;
}

// Load a 16(rows) x 32(K) bf16 WMMA fragment, element (i,k) = src[(row0+i)*stride + k0 + k].
// Matches ISA 7.12.2 "16-bit A-Matrix 16x32": lane&15 = row, lane>>4 selects K half,
// VGPRs 0..3 hold K = 0..7 (+8*half), VGPRs 4..7 hold K = 16..23 (+8*half).
// Also used for B fragments when the source tile is stored K-contiguous per column
// (i.e. already transposed), since B's layout mirrors A's with N in place of M.
__device__ inline v16bf frag_ld(const __bf16* __restrict__ s, int row0, int k0, int stride) {
  const int lane = threadIdx.x & 31;
  const int r  = row0 + (lane & 15);
  const int kh = (lane >> 4) * 8;
  const __bf16* p = s + r * stride + k0 + kh;
  v16bf f;
#pragma unroll
  for (int j = 0; j < 8; ++j) f[j] = p[j];          // K = kh + 0..7
  const __bf16* q = p + 16;
#pragma unroll
  for (int j = 0; j < 8; ++j) f[8 + j] = q[j];      // K = kh + 16..23
  return f;
}

__device__ inline v8f wmma_bf16(v16bf a, v16bf b, v8f c) {
  return __builtin_amdgcn_wmma_f32_16x16x32_bf16(false, a, false, b, (short)0, c, false, false);
}

struct SharedGemm {
  union {
    struct { __bf16 a[128 * KSTR]; __bf16 b[128 * KSTR]; } stage;  // 36 KB
    __bf16 tile[128 * TSTR];                                        // 34 KB
  };
};

// ---------------------------------------------------------------------------
// Kernel 1: Q/K/V projection (Y = X @ W^T) fused with RMSNorm + rotary.
// grid = (S/128, NHEADS, 3). Each block produces a 128x128 tile that is
// exactly one head's columns, so norm/rotary are block-local.
// ---------------------------------------------------------------------------
__global__ __launch_bounds__(256) void qkv_rot_kernel(
    const float* __restrict__ X,  const float* __restrict__ Wq,
    const float* __restrict__ Wk, const float* __restrict__ Wv,
    const float* __restrict__ freqs, const float* __restrict__ gq,
    const float* __restrict__ gk, __bf16* __restrict__ out_qkv /* [3][H][S][Hd] */) {
  __shared__ SharedGemm sm;
  const int m0    = blockIdx.x * 128;
  const int h     = blockIdx.y;
  const int which = blockIdx.z;                 // 0=q 1=k 2=v
  const float* W  = (which == 0) ? Wq : ((which == 1) ? Wk : Wv);
  const int n0    = h * HDIM;
  const int t     = threadIdx.x;
  const int wave  = t >> 5;
  const int lane  = t & 31;

  v8f acc[8];
#pragma unroll
  for (int i = 0; i < 8; ++i) acc[i] = vzero8();

  for (int k0 = 0; k0 < DMODEL; k0 += KB) {
    __syncthreads();
#pragma unroll
    for (int j = 0; j < 8; ++j) {               // 2048 float4 chunks / 256 threads
      const int idx = t + j * 256;
      const int row = idx >> 4;                 // 16 float4 per 64-wide row
      const int c4  = (idx & 15) * 4;
      const float4 xa = *(const float4*)(X + (size_t)(m0 + row) * DMODEL + k0 + c4);
      __bf16* da = &sm.stage.a[row * KSTR + c4];
      da[0] = (__bf16)xa.x; da[1] = (__bf16)xa.y; da[2] = (__bf16)xa.z; da[3] = (__bf16)xa.w;
      const float4 xb = *(const float4*)(W + (size_t)(n0 + row) * DMODEL + k0 + c4);
      __bf16* db = &sm.stage.b[row * KSTR + c4];
      db[0] = (__bf16)xb.x; db[1] = (__bf16)xb.y; db[2] = (__bf16)xb.z; db[3] = (__bf16)xb.w;
    }
    __syncthreads();
#pragma unroll
    for (int ks = 0; ks < KB; ks += 32) {
      const v16bf a = frag_ld(sm.stage.a, wave * 16, ks, KSTR);
#pragma unroll
      for (int n = 0; n < 8; ++n) {
        const v16bf b = frag_ld(sm.stage.b, n * 16, ks, KSTR);   // B[k][n] = W[n0+n][k]
        acc[n] = wmma_bf16(a, b, acc[n]);
      }
    }
  }
  __syncthreads();
  // Spill C tile to LDS so norm/rotary can see whole rows.
  {
    const int nlo = lane & 15;
    const int mhi = (lane >> 4) * 8;
#pragma unroll
    for (int n = 0; n < 8; ++n)
#pragma unroll
      for (int r = 0; r < 8; ++r)
        sm.tile[(wave * 16 + mhi + r) * TSTR + n * 16 + nlo] = (__bf16)acc[n][r];
  }
  __syncthreads();

  if (t < 128) {                                 // one thread per output row
    const int srow = m0 + t;
    __bf16* dst = out_qkv + ((size_t)which * NHEADS + h) * (size_t)S_LEN * HDIM
                          + (size_t)srow * HDIM;
    if (which == 2) {
#pragma unroll
      for (int i = 0; i < HDIM; ++i) dst[i] = sm.tile[t * TSTR + i];
    } else {
      const float* g = (which == 0) ? gq : gk;
      float ss = 0.f;
#pragma unroll
      for (int i = 0; i < HDIM; ++i) {
        const float xi = (float)sm.tile[t * TSTR + i];
        ss += xi * xi;
      }
      const float rinv = rsqrtf(ss * (1.0f / HDIM) + 1e-5f);
#pragma unroll
      for (int i = 0; i < HDIM; ++i) {
        const float xi = (float)sm.tile[t * TSTR + i] * rinv * g[i];
        const int   ip = (i < 64) ? (i + 64) : (i - 64);
        float xo = (float)sm.tile[t * TSTR + ip] * rinv * g[ip];
        if (i < 64) xo = -xo;
        const float f = freqs[(size_t)srow * HDIM + i];
        dst[i] = (__bf16)(xi * __cosf(f) + xo * __sinf(f));
      }
    }
  }
}

// ---------------------------------------------------------------------------
// Kernel 2: flash attention per (query block, head). 128 query rows / block,
// 8 waves each owning 16 full rows -> softmax reductions stay inside a wave
// (16-lane shfl_xor halves).
// ---------------------------------------------------------------------------
struct SharedAttn {
  __bf16 q[128 * TSTR];        // pre-scaled Q tile
  __bf16 k[128 * TSTR];        // K block, [key][hd]
  __bf16 vt[128 * TSTR];       // V block transposed, [hd][key]
  __bf16 p[8][16 * TSTR];      // per-wave bf16 P tiles
};

__global__ __launch_bounds__(256) void attn_kernel(
    const __bf16* __restrict__ qkv, __bf16* __restrict__ attn_out /* [S][H*Hd] */) {
  __shared__ SharedAttn sm;
  const int q0   = blockIdx.x * 128;
  const int h    = blockIdx.y;
  const int t    = threadIdx.x;
  const int wave = t >> 5;
  const int lane = t & 31;
  const size_t hs = (size_t)S_LEN * HDIM;
  const __bf16* Q = qkv + (size_t)h * hs;
  const __bf16* K = qkv + ((size_t)NHEADS + h) * hs;
  const __bf16* V = qkv + ((size_t)2 * NHEADS + h) * hs;

  const float qscale = 0.08838834764831845f;     // 1/sqrt(128)
#pragma unroll
  for (int j = 0; j < 16; ++j) {                 // 4096 4-elem chunks / 256 threads
    const int idx = t + j * 256;
    const int row = idx >> 5;
    const int c4  = (idx & 31) * 4;
    const __bf16* src = Q + (size_t)(q0 + row) * HDIM + c4;
    __bf16* d = &sm.q[row * TSTR + c4];
#pragma unroll
    for (int e = 0; e < 4; ++e) d[e] = (__bf16)((float)src[e] * qscale);
  }

  v8f o[8];
  float mrow[8], lrow[8];
#pragma unroll
  for (int r = 0; r < 8; ++r) { o[r] = vzero8(); mrow[r] = -3.0e38f; lrow[r] = 0.f; }

  for (int kb = 0; kb < S_LEN; kb += 128) {
    __syncthreads();
#pragma unroll
    for (int j = 0; j < 16; ++j) {
      const int idx = t + j * 256;
      const int row = idx >> 5;
      const int c4  = (idx & 31) * 4;
      const __bf16* sk = K + (size_t)(kb + row) * HDIM + c4;
      __bf16* d = &sm.k[row * TSTR + c4];
      d[0] = sk[0]; d[1] = sk[1]; d[2] = sk[2]; d[3] = sk[3];
      const __bf16* sv = V + (size_t)(kb + row) * HDIM + c4;
#pragma unroll
      for (int e = 0; e < 4; ++e) sm.vt[(c4 + e) * TSTR + row] = sv[e];
    }
    __syncthreads();

    // S = Q @ K^T for this wave's 16 rows x 128 keys
    v8f sacc[8];
#pragma unroll
    for (int n = 0; n < 8; ++n) sacc[n] = vzero8();
    v16bf aq[4];
#pragma unroll
    for (int ks = 0; ks < 4; ++ks) aq[ks] = frag_ld(sm.q, wave * 16, ks * 32, TSTR);
#pragma unroll
    for (int n = 0; n < 8; ++n)
#pragma unroll
      for (int ks = 0; ks < 4; ++ks) {
        const v16bf bk = frag_ld(sm.k, n * 16, ks * 32, TSTR);  // B[k][n] = K[n][k]
        sacc[n] = wmma_bf16(aq[ks], bk, sacc[n]);
      }

    // Online softmax (per-row stats; rows split across 16-lane halves)
    float corr[8];
#pragma unroll
    for (int r = 0; r < 8; ++r) {
      float m = mrow[r];
#pragma unroll
      for (int n = 0; n < 8; ++n) m = fmaxf(m, sacc[n][r]);
#pragma unroll
      for (int off = 1; off < 16; off <<= 1) m = fmaxf(m, __shfl_xor(m, off));
      corr[r] = __expf(mrow[r] - m);
      mrow[r] = m;
    }
    float rs[8];
#pragma unroll
    for (int r = 0; r < 8; ++r) rs[r] = 0.f;
#pragma unroll
    for (int n = 0; n < 8; ++n)
#pragma unroll
      for (int r = 0; r < 8; ++r) {
        const float p = __expf(sacc[n][r] - mrow[r]);
        sacc[n][r] = p;
        rs[r] += p;
      }
#pragma unroll
    for (int r = 0; r < 8; ++r) {
#pragma unroll
      for (int off = 1; off < 16; off <<= 1) rs[r] += __shfl_xor(rs[r], off);
      lrow[r] = lrow[r] * corr[r] + rs[r];
#pragma unroll
      for (int n = 0; n < 8; ++n) o[n][r] *= corr[r];
    }

    // Round P to bf16 in this wave's LDS region, then feed WMMA as A.
    {
      const int nlo = lane & 15;
      const int mhi = (lane >> 4) * 8;
#pragma unroll
      for (int n = 0; n < 8; ++n)
#pragma unroll
        for (int r = 0; r < 8; ++r)
          sm.p[wave][(mhi + r) * TSTR + n * 16 + nlo] = (__bf16)sacc[n][r];
    }
    __syncthreads();

    // O += P @ V   (A = P[16 x 128keys], B[k][n] = V[k][n] = vt[n][k])
    v16bf ap[4];
#pragma unroll
    for (int ks = 0; ks < 4; ++ks) ap[ks] = frag_ld(sm.p[wave], 0, ks * 32, TSTR);
#pragma unroll
    for (int n = 0; n < 8; ++n)
#pragma unroll
      for (int ks = 0; ks < 4; ++ks) {
        const v16bf bv = frag_ld(sm.vt, n * 16, ks * 32, TSTR);
        o[n] = wmma_bf16(ap[ks], bv, o[n]);
      }
  }

  // Normalize and store attention output (bf16, [s][h*Hd]).
  {
    const int nlo = lane & 15;
    const int mhi = (lane >> 4) * 8;
#pragma unroll
    for (int r = 0; r < 8; ++r) {
      const float inv = 1.f / lrow[r];
      const int srow  = q0 + wave * 16 + mhi + r;
      __bf16* dst = attn_out + (size_t)srow * DMODEL + h * HDIM;
#pragma unroll
      for (int n = 0; n < 8; ++n) dst[n * 16 + nlo] = (__bf16)(o[n][r] * inv);
    }
  }
}

// ---------------------------------------------------------------------------
// Kernel 3: output projection  Y = Attn @ Wo^T + bo  (fp32 out)
// ---------------------------------------------------------------------------
__global__ __launch_bounds__(256) void oproj_kernel(
    const __bf16* __restrict__ A, const float* __restrict__ Wo,
    const float* __restrict__ bo, float* __restrict__ Y) {
  __shared__ SharedGemm sm;
  const int m0   = blockIdx.x * 128;
  const int n0   = blockIdx.y * 128;
  const int t    = threadIdx.x;
  const int wave = t >> 5;
  const int lane = t & 31;

  v8f acc[8];
#pragma unroll
  for (int i = 0; i < 8; ++i) acc[i] = vzero8();

  for (int k0 = 0; k0 < DMODEL; k0 += KB) {
    __syncthreads();
#pragma unroll
    for (int j = 0; j < 8; ++j) {
      const int idx = t + j * 256;
      const int row = idx >> 4;
      const int c4  = (idx & 15) * 4;
      const __bf16* sa = A + (size_t)(m0 + row) * DMODEL + k0 + c4;
      __bf16* da = &sm.stage.a[row * KSTR + c4];
      da[0] = sa[0]; da[1] = sa[1]; da[2] = sa[2]; da[3] = sa[3];
      const float4 xb = *(const float4*)(Wo + (size_t)(n0 + row) * DMODEL + k0 + c4);
      __bf16* db = &sm.stage.b[row * KSTR + c4];
      db[0] = (__bf16)xb.x; db[1] = (__bf16)xb.y; db[2] = (__bf16)xb.z; db[3] = (__bf16)xb.w;
    }
    __syncthreads();
#pragma unroll
    for (int ks = 0; ks < KB; ks += 32) {
      const v16bf a = frag_ld(sm.stage.a, wave * 16, ks, KSTR);
#pragma unroll
      for (int n = 0; n < 8; ++n) {
        const v16bf b = frag_ld(sm.stage.b, n * 16, ks, KSTR);
        acc[n] = wmma_bf16(a, b, acc[n]);
      }
    }
  }
  const int nlo = lane & 15;
  const int mhi = (lane >> 4) * 8;
#pragma unroll
  for (int n = 0; n < 8; ++n) {
    const float bias = bo[n0 + n * 16 + nlo];
#pragma unroll
    for (int r = 0; r < 8; ++r) {
      const int row = m0 + wave * 16 + mhi + r;
      Y[(size_t)row * DMODEL + n0 + n * 16 + nlo] = acc[n][r] + bias;
    }
  }
}

// ---------------------------------------------------------------------------
extern "C" void kernel_launch(void* const* d_in, const int* in_sizes, int n_in,
                              void* d_out, int out_size, void* d_ws, size_t ws_size,
                              hipStream_t stream) {
  const float* hs    = (const float*)d_in[0];
  const float* freqs = (const float*)d_in[1];
  const float* Wq    = (const float*)d_in[2];
  const float* Wk    = (const float*)d_in[3];
  const float* Wv    = (const float*)d_in[4];
  const float* Wo    = (const float*)d_in[5];
  const float* bo    = (const float*)d_in[6];
  const float* gq    = (const float*)d_in[7];
  const float* gk    = (const float*)d_in[8];

  __bf16* qkv  = (__bf16*)d_ws;                                   // 3*16*4096*128 bf16 = 48 MB
  __bf16* attn = qkv + (size_t)3 * NHEADS * S_LEN * HDIM;         // 4096*2048 bf16   = 16 MB
  float*  out  = (float*)d_out;

  dim3 blk(256);
  qkv_rot_kernel<<<dim3(S_LEN / 128, NHEADS, 3), blk, 0, stream>>>(hs, Wq, Wk, Wv, freqs, gq, gk, qkv);
  attn_kernel<<<dim3(S_LEN / 128, NHEADS), blk, 0, stream>>>(qkv, attn);
  oproj_kernel<<<dim3(S_LEN / 128, DMODEL / 128), blk, 0, stream>>>(attn, Wo, bo, out);
}
